// MxDNADeformableConvBlock_11690900980377
// MI455X (gfx1250) — matrix operand order repaired and checked by reading, not compiled, not run on hardware
//
#include <hip/hip_runtime.h>
#include <hip/hip_bf16.h>
#include <math.h>

typedef __attribute__((ext_vector_type(16))) __bf16 v16bf;
typedef __attribute__((ext_vector_type(8)))  __bf16 v8bf;
typedef __attribute__((ext_vector_type(4)))  __bf16 v4bf;
typedef __attribute__((ext_vector_type(8)))  float  v8f;

#define NTOK  8192          // N
#define DDIM  256           // D
#define KC    768           // contraction = K*D
#define MT    64            // tokens per block
#define APAD  8
#define ASTR  (KC + APAD)   // 776 bf16 per LDS row -> stride 388 dwords (== 4 mod 64 banks)

// --- Kernel 0: repack weight (D,E,K) f32 -> Wt[d][k*256+e] bf16 (column-major for B frags) ---
__global__ void wt_pack_kernel(const float* __restrict__ w, __bf16* __restrict__ wt) {
    int idx = blockIdx.x * 256 + threadIdx.x;   // 0..196607 == 256*768
    int d = idx / KC;
    int c = idx - d * KC;
    int k = c >> 8;          // tap
    int e = c & 255;         // input channel
    wt[idx] = (__bf16)w[(d * 256 + e) * 3 + k];
}

// --- Kernel 1: fused offsets + sigmoid modulation + bilinear gather + bf16 WMMA GEMM ---
__global__ __launch_bounds__(256) void deform_conv_kernel(
    const float* __restrict__ x,      // (B, N, D) f32
    const __bf16* __restrict__ wt,    // (D, KC) bf16, column-major B matrix
    const float* __restrict__ ow,     // (K, D) offset weights
    const float* __restrict__ mw,     // (K, D) modulation weights
    float* __restrict__ out)          // (B, N, D) f32
{
    __shared__ __bf16 sA[MT * ASTR];  // 99,328 B : bf16 "modulated" A-tile
    __shared__ float  sOw[KC];        // 3 KB
    __shared__ float  sMw[KC];        // 3 KB

    const int tid = threadIdx.x;

    // stage small weights into LDS
    for (int i = tid; i < KC; i += 256) { sOw[i] = ow[i]; sMw[i] = mw[i]; }
    __syncthreads();

    // ---------------- Stage 1: build A-tile (64 tokens x 768 bf16) ----------------
    {
        const int tt = tid >> 2;            // token in tile (0..63)
        const int q  = tid & 3;             // quarter of D (64 channels each)
        const int g  = blockIdx.x * MT + tt;
        const int b  = g >> 13;             // N = 8192
        const int i  = g & (NTOK - 1);
        const float* xb = x + (((size_t)b << 13) * DDIM);
        const float* xt = xb + (size_t)i * DDIM + q * 64;

        float so0 = 0.f, so1 = 0.f, so2 = 0.f;
        float sm0 = 0.f, sm1 = 0.f, sm2 = 0.f;
        const int qo = q * 64;
        #pragma unroll 4
        for (int e = 0; e < 64; e += 4) {
            float4 xv = *(const float4*)(xt + e);
            float4 a0 = *(const float4*)(&sOw[0 * 256 + qo + e]);
            float4 a1 = *(const float4*)(&sOw[1 * 256 + qo + e]);
            float4 a2 = *(const float4*)(&sOw[2 * 256 + qo + e]);
            float4 m0 = *(const float4*)(&sMw[0 * 256 + qo + e]);
            float4 m1 = *(const float4*)(&sMw[1 * 256 + qo + e]);
            float4 m2 = *(const float4*)(&sMw[2 * 256 + qo + e]);
            so0 += xv.x * a0.x + xv.y * a0.y + xv.z * a0.z + xv.w * a0.w;
            so1 += xv.x * a1.x + xv.y * a1.y + xv.z * a1.z + xv.w * a1.w;
            so2 += xv.x * a2.x + xv.y * a2.y + xv.z * a2.z + xv.w * a2.w;
            sm0 += xv.x * m0.x + xv.y * m0.y + xv.z * m0.z + xv.w * m0.w;
            sm1 += xv.x * m1.x + xv.y * m1.y + xv.z * m1.z + xv.w * m1.w;
            sm2 += xv.x * m2.x + xv.y * m2.y + xv.z * m2.z + xv.w * m2.w;
        }
        // butterfly reduce over the 4 lanes of this token (all lanes end with full sum)
        #pragma unroll
        for (int m = 1; m <= 2; m <<= 1) {
            so0 += __shfl_xor(so0, m); so1 += __shfl_xor(so1, m); so2 += __shfl_xor(so2, m);
            sm0 += __shfl_xor(sm0, m); sm1 += __shfl_xor(sm1, m); sm2 += __shfl_xor(sm2, m);
        }
        const float soff[3] = { so0, so1, so2 };
        const float smod[3] = { sm0, sm1, sm2 };

        #pragma unroll
        for (int k = 0; k < 3; ++k) {
            float modv  = 1.0f / (1.0f + expf(-smod[k]));
            float pos   = (float)(i + k - 1) + soff[k];
            bool  valid = (pos >= 0.0f) && (pos < (float)NTOK);
            float ff    = floorf(pos);
            ff = fminf(fmaxf(ff, 0.0f), (float)(NTOK - 1));
            int   fi = (int)ff;
            int   ci = fi + 1; if (ci > NTOK - 1) ci = NTOK - 1;
            float wc = pos - ff;
            float wf = 1.0f - wc;
            float mf  = valid ? modv : 0.0f;
            float wfm = wf * mf, wcm = wc * mf;
            const float* pf = xb + (size_t)fi * DDIM + qo;
            const float* pc = xb + (size_t)ci * DDIM + qo;
            __bf16* dst = &sA[tt * ASTR + k * 256 + qo];
            #pragma unroll 4
            for (int e = 0; e < 64; e += 4) {
                float4 vf = *(const float4*)(pf + e);
                float4 vc = *(const float4*)(pc + e);
                v4bf o;
                o[0] = (__bf16)(vf.x * wfm + vc.x * wcm);
                o[1] = (__bf16)(vf.y * wfm + vc.y * wcm);
                o[2] = (__bf16)(vf.z * wfm + vc.z * wcm);
                o[3] = (__bf16)(vf.w * wfm + vc.w * wcm);
                *(v4bf*)(dst + e) = o;
            }
        }
    }
    __syncthreads();

    // ---------------- Stage 2: (64 x 768) x (768 x 256) bf16 WMMA GEMM ----------------
    {
        const int w     = tid >> 5;         // wave 0..7
        const int lane  = tid & 31;
        const int l16   = lane & 15;
        const int lhalf = lane >> 4;        // 0 or 1
        const int wmrow = (w & 1) * 32;     // 2 M-groups of 32 rows
        const int wncol = (w >> 1) * 64;    // 4 N-groups of 64 cols

        v8f acc[2][4];
        #pragma unroll
        for (int mt = 0; mt < 2; ++mt)
            #pragma unroll
            for (int nt = 0; nt < 4; ++nt)
                acc[mt][nt] = (v8f){0.f,0.f,0.f,0.f,0.f,0.f,0.f,0.f};

        for (int kk = 0; kk < KC; kk += 32) {
            // A fragments (16-bit A layout: lanes 0-15 K 0-7/16-23, lanes 16-31 K 8-15/24-31)
            v16bf afrag[2];
            #pragma unroll
            for (int mt = 0; mt < 2; ++mt) {
                int row = wmrow + mt * 16 + l16;
                const __bf16* ap = &sA[row * ASTR + kk + lhalf * 8];
                v8bf lo = *(const v8bf*)ap;           // ds_load_b128
                v8bf hi = *(const v8bf*)(ap + 16);    // ds_load_b128
                afrag[mt] = __builtin_shufflevector(lo, hi,
                    0,1,2,3,4,5,6,7,8,9,10,11,12,13,14,15);
            }
            #pragma unroll
            for (int nt = 0; nt < 4; ++nt) {
                // B fragment (32x16): lanes 0-15 K 0-7/16-23, lanes 16-31 K 8-15/24-31, column l16
                int col = wncol + nt * 16 + l16;
                const __bf16* bp = wt + (size_t)col * KC + kk + lhalf * 8;
                v8bf lo = *(const v8bf*)bp;           // global_load_b128
                v8bf hi = *(const v8bf*)(bp + 16);    // global_load_b128
                v16bf bfrag = __builtin_shufflevector(lo, hi,
                    0,1,2,3,4,5,6,7,8,9,10,11,12,13,14,15);
                #pragma unroll
                for (int mt = 0; mt < 2; ++mt) {
                    acc[mt][nt] = __builtin_amdgcn_wmma_f32_16x16x32_bf16(
                        false, afrag[mt], false, bfrag,
                        (short)0, acc[mt][nt], false, false);
                }
            }
        }

        // Epilogue: C/D layout — VGPR r: lanes 0-15 M=r, lanes 16-31 M=r+8; N=l16
        const size_t tok0 = (size_t)blockIdx.x * MT;
        #pragma unroll
        for (int mt = 0; mt < 2; ++mt) {
            #pragma unroll
            for (int nt = 0; nt < 4; ++nt) {
                int col = wncol + nt * 16 + l16;
                #pragma unroll
                for (int r = 0; r < 8; ++r) {
                    size_t row = tok0 + wmrow + mt * 16 + r + lhalf * 8;
                    out[row * DDIM + col] = acc[mt][nt][r];
                }
            }
        }
    }
}

extern "C" void kernel_launch(void* const* d_in, const int* in_sizes, int n_in,
                              void* d_out, int out_size, void* d_ws, size_t ws_size,
                              hipStream_t stream) {
    const float* x   = (const float*)d_in[0];   // basic_units (B,N,D)
    const float* ow  = (const float*)d_in[1];   // offset_w (K,D)
    const float* mw  = (const float*)d_in[2];   // mod_w (K,D)
    const float* wgt = (const float*)d_in[3];   // weight (D,D,K)
    float* out = (float*)d_out;

    __bf16* wt = (__bf16*)d_ws;                 // 256*768 bf16 = 384 KB

    wt_pack_kernel<<<768, 256, 0, stream>>>(wgt, wt);
    deform_conv_kernel<<<(8 * NTOK) / MT, 256, 0, stream>>>(x, wt, ow, mw, out);
}
